// ManualFlashAttention_68118181314685
// MI455X (gfx1250) — matrix operand order repaired and checked by reading, not compiled, not run on hardware
//
#include <hip/hip_runtime.h>
#include <hip/hip_bf16.h>

typedef _Float16 v16h __attribute__((ext_vector_type(16)));
typedef _Float16 v8h  __attribute__((ext_vector_type(8)));
typedef _Float16 v4h  __attribute__((ext_vector_type(4)));
typedef float    v8f  __attribute__((ext_vector_type(8)));
typedef int      v4i  __attribute__((ext_vector_type(4)));

#define DIMC 1024
#define NHEAD 16
#define HDIM 64
#define NTOK 4096   // B*T = 2*2048
#define SEQ 2048

// ---------------------------------------------------------------------------
// CDNA5 async global->LDS copy (ASYNCcnt-tracked), with sync fallback
// builtin signature (probe-confirmed): (v4i as1*, v4i as3*, imm, imm)
// ---------------------------------------------------------------------------
#if defined(__AMDGCN__) && __has_builtin(__builtin_amdgcn_global_load_async_to_lds_b128)
#define HAVE_ASYNC 1
#define ASYNC_CP16(gsrc, ldst)                                                  \
    __builtin_amdgcn_global_load_async_to_lds_b128(                             \
        (__attribute__((address_space(1))) v4i*)(gsrc),                         \
        (__attribute__((address_space(3))) v4i*)(ldst), 0, 0)
#define WAIT_ASYNC(n) asm volatile("s_wait_asynccnt %0" ::"i"(n) : "memory")
#else
#define HAVE_ASYNC 0
#define ASYNC_CP16(gsrc, ldst) (*(v8h*)(ldst) = *(const v8h*)(gsrc))
#define WAIT_ASYNC(n) do { } while (0)
#endif

// ---------------------------------------------------------------------------
// helpers
// ---------------------------------------------------------------------------
__device__ __forceinline__ v16h make_frag(v8h g0, v8h g1) {
    v16h r;
#pragma unroll
    for (int i = 0; i < 8; i++) { r[i] = g0[i]; r[i + 8] = g1[i]; }
    return r;
}

__device__ __forceinline__ v4h cvt4(float4 v) {
    v4h h; h[0] = (_Float16)v.x; h[1] = (_Float16)v.y;
    h[2] = (_Float16)v.z; h[3] = (_Float16)v.w;
    return h;
}

__device__ __forceinline__ float redmax16(float v) {
#pragma unroll
    for (int m = 1; m < 16; m <<= 1) v = fmaxf(v, __shfl_xor(v, m, 32));
    return v;
}
__device__ __forceinline__ float redsum16(float v) {
#pragma unroll
    for (int m = 1; m < 16; m <<= 1) v += __shfl_xor(v, m, 32);
    return v;
}

#define WMMA_F16(A, B, C) \
    __builtin_amdgcn_wmma_f32_16x16x32_f16(false, (A), false, (B), (short)0, (C), false, false)

// A-frag (16x32 f16, MxK): lane m = lane&15, kh = lane>>4.
//   e -> k = kh*16 + e
// B-frag (32x16 f16, KxN): lane n = lane&15, kh = lane>>4; e -> k = kh*16 + e
// C/D (16x16 f32): lane n = lane&15; vgpr r -> m = (lane>>4)*8 + r

// ---------------------------------------------------------------------------
// Kernel 1: fused QKV projection  (out[n,j] = sum_c x[n,c]*W[j,c] + b[j])
// M=4096 tokens, N=3072 features, K=1024.  Register-pipelined staging:
// next K-slice's global loads are in flight during current tile's WMMAs.
// Scatter f16: Q,K -> [B,H,T,D]; V -> [B,H,D,T] (pre-transposed for attention).
// Scale 1/sqrt(64) folded into Q.
// ---------------------------------------------------------------------------
#define AS_STR 40
#define BS_STR 40

__global__ __launch_bounds__(256) void qkv_proj_kernel(
    const float* __restrict__ x, const float* __restrict__ W,
    const float* __restrict__ bias,
    _Float16* __restrict__ Qb, _Float16* __restrict__ Kb, _Float16* __restrict__ Vb)
{
    __shared__ _Float16 As[128 * AS_STR] __attribute__((aligned(16)));
    __shared__ _Float16 Bs[64 * BS_STR] __attribute__((aligned(16)));

    const int tid  = threadIdx.x;
    const int lane = tid & 31, wave = tid >> 5;
    const int wm = wave >> 1, wn = wave & 1;
    const int l16 = lane & 15, kh = lane >> 4;
    const int Mbase = blockIdx.y * 128;
    const int Nbase = blockIdx.x * 64;

    float4 gA[4], gB[2];
    auto load_tiles = [&](int kc) {
#pragma unroll
        for (int i = 0; i < 4; i++) {   // A: 128x32 f32, 1024 float4, 4/thread
            int f = tid + i * 256;
            int row = f >> 3, c4 = (f & 7) * 4;
            gA[i] = *(const float4*)&x[(size_t)(Mbase + row) * DIMC + kc + c4];
        }
#pragma unroll
        for (int i = 0; i < 2; i++) {   // B: 64x32 f32, 512 float4, 2/thread
            int f = tid + i * 256;
            int row = f >> 3, c4 = (f & 7) * 4;
            gB[i] = *(const float4*)&W[(size_t)(Nbase + row) * DIMC + kc + c4];
        }
    };
    auto store_tiles = [&]() {
#pragma unroll
        for (int i = 0; i < 4; i++) {
            int f = tid + i * 256;
            int row = f >> 3, c4 = (f & 7) * 4;
            *(v4h*)&As[row * AS_STR + c4] = cvt4(gA[i]);
        }
#pragma unroll
        for (int i = 0; i < 2; i++) {
            int f = tid + i * 256;
            int row = f >> 3, c4 = (f & 7) * 4;
            *(v4h*)&Bs[row * BS_STR + c4] = cvt4(gB[i]);
        }
    };

    v8f acc[2][2] = {};

    load_tiles(0);
    for (int kc = 0; kc < DIMC; kc += 32) {
        __syncthreads();               // previous tile's reads complete
        store_tiles();                 // f32->f16 convert into LDS
        __syncthreads();               // tile visible to all waves
        if (kc + 32 < DIMC) load_tiles(kc + 32);   // overlap with WMMAs

        v16h af[2], bf[2];
#pragma unroll
        for (int a = 0; a < 2; a++) {
            const _Float16* ar = &As[(wm * 32 + a * 16 + l16) * AS_STR];
            af[a] = make_frag(*(const v8h*)&ar[kh * 16], *(const v8h*)&ar[kh * 16 + 8]);
        }
#pragma unroll
        for (int bi = 0; bi < 2; bi++) {
            const _Float16* br = &Bs[(wn * 32 + bi * 16 + l16) * BS_STR + kh * 16];
            bf[bi] = make_frag(*(const v8h*)&br[0], *(const v8h*)&br[8]);
        }
#pragma unroll
        for (int a = 0; a < 2; a++)
#pragma unroll
            for (int bi = 0; bi < 2; bi++)
                acc[a][bi] = WMMA_F16(af[a], bf[bi], acc[a][bi]);
    }

    // epilogue: + bias, scatter f16 (scale folded into Q; V stored transposed)
#pragma unroll
    for (int bi = 0; bi < 2; bi++) {
        int j = Nbase + wn * 32 + bi * 16 + l16;         // 0..3071
        float bj = bias[j];
        int which = j >> 10, rem = j & 1023;
        int hh = rem >> 6, dd = rem & 63;
        _Float16* dst = (which == 0) ? Qb : (which == 1) ? Kb : Vb;
        float sc = (which == 0) ? 0.125f : 1.0f;         // 1/sqrt(64)
#pragma unroll
        for (int a = 0; a < 2; a++) {
#pragma unroll
            for (int r = 0; r < 8; r++) {
                int gm = Mbase + wm * 32 + a * 16 + kh * 8 + r;  // token 0..4095
                int bb = gm >> 11, tt = gm & 2047;
                float val = (acc[a][bi][r] + bj) * sc;
                size_t idx;
                if (which == 2)   // V^T: [B,H,D,T]
                    idx = ((size_t)(bb * NHEAD + hh) * HDIM + dd) * SEQ + tt;
                else              // Q,K: [B,H,T,D]
                    idx = ((size_t)(bb * NHEAD + hh) * SEQ + tt) * HDIM + dd;
                dst[idx] = (_Float16)val;
            }
        }
    }
}

// ---------------------------------------------------------------------------
// Kernel 2: flash attention, one (b,h) per blockIdx.y, 128 queries per WG,
// 16 queries per wave. 64-key blocks double-buffered in LDS via async copies.
// ---------------------------------------------------------------------------
#define KSTR 72

__global__ __launch_bounds__(256) void flash_attn_kernel(
    const _Float16* __restrict__ Qb, const _Float16* __restrict__ Kb,
    const _Float16* __restrict__ Vb, _Float16* __restrict__ Ob)
{
    __shared__ _Float16 Kl[2][64 * KSTR] __attribute__((aligned(16)));  // [key][d]
    __shared__ _Float16 Vl[2][64 * KSTR] __attribute__((aligned(16)));  // [d][key]
    __shared__ _Float16 Pl[8 * 16 * KSTR] __attribute__((aligned(16)));

    const int tid  = threadIdx.x;
    const int lane = tid & 31, wave = tid >> 5;
    const int l16 = lane & 15, kh = lane >> 4;
    const int bh = blockIdx.y;                       // b*16+h
    const int qbase = blockIdx.x * 128 + wave * 16;
    const size_t base = (size_t)bh * SEQ * HDIM;     // same extent for K and V^T

    // Q fragments live in registers for the whole kernel (d split into 2x32)
    v16h aq[2];
    {
        const _Float16* qr = &Qb[base + (size_t)(qbase + l16) * HDIM];
#pragma unroll
        for (int c = 0; c < 2; c++)
            aq[c] = make_frag(*(const v8h*)&qr[c * 32 + kh * 16],
                              *(const v8h*)&qr[c * 32 + kh * 16 + 8]);
    }

    float mrow[8], lrow[8];
    v8f oacc[4] = {};
#pragma unroll
    for (int r = 0; r < 8; r++) { mrow[r] = -1e30f; lrow[r] = 0.0f; }

    _Float16* Pw = &Pl[wave * 16 * KSTR];

    // issue one 64-key block of K and V^T into LDS buffer `buf` (4 async/thread)
    auto stage = [&](int kb, int buf) {
#pragma unroll
        for (int i = 0; i < 2; i++) {
            int f = tid + i * 256;                    // 512 16B chunks per array
            int r = f >> 3, cg = (f & 7) * 8;
            ASYNC_CP16(&Kb[base + (size_t)(kb + r) * HDIM + cg],
                       &Kl[buf][r * KSTR + cg]);
            ASYNC_CP16(&Vb[base + (size_t)r * SEQ + kb + cg],
                       &Vl[buf][r * KSTR + cg]);
        }
    };

    stage(0, 0);
    for (int it = 0; it < SEQ / 64; it++) {
        const int cur = it & 1;
        if (it + 1 < SEQ / 64) {
            stage((it + 1) * 64, (it + 1) & 1);  // prefetch next block
            WAIT_ASYNC(4);                       // current block's 4 copies done
        } else {
            WAIT_ASYNC(0);
        }
        __syncthreads();

        // S = Q @ K^T  (scale already folded into Q)
        v8f s[4];
#pragma unroll
        for (int j = 0; j < 4; j++) {
            v8f c = {};
#pragma unroll
            for (int ch = 0; ch < 2; ch++) {
                const _Float16* kr = &Kl[cur][(j * 16 + l16) * KSTR + ch * 32 + kh * 16];
                v16h bf = make_frag(*(const v8h*)&kr[0], *(const v8h*)&kr[8]);
                c = WMMA_F16(aq[ch], bf, c);
            }
            s[j] = c;
        }

        // online softmax update + stash P (f16) into per-wave LDS scratch
#pragma unroll
        for (int r = 0; r < 8; r++) {
            float lm = fmaxf(fmaxf(s[0][r], s[1][r]), fmaxf(s[2][r], s[3][r]));
            lm = redmax16(lm);
            float mnew  = fmaxf(mrow[r], lm);
            float alpha = __expf(mrow[r] - mnew);
            float psum = 0.0f;
#pragma unroll
            for (int j = 0; j < 4; j++) {
                float p = __expf(s[j][r] - mnew);
                s[j][r] = p;
                psum += p;
            }
            psum = redsum16(psum);
            lrow[r] = lrow[r] * alpha + psum;
            mrow[r] = mnew;
#pragma unroll
            for (int jd = 0; jd < 4; jd++) oacc[jd][r] *= alpha;
            int m = kh * 8 + r;
#pragma unroll
            for (int j = 0; j < 4; j++)
                Pw[m * KSTR + j * 16 + l16] = (_Float16)s[j][r];
        }
        // per-wave LDS RAW: DS ops of a wave are in-order; wait for them
        asm volatile("s_wait_dscnt 0x0" ::: "memory");

        // O += P @ V   (A = P from LDS, B rows are d, cols are key)
        v16h ap[2];
        {
            const _Float16* pr = &Pw[l16 * KSTR];
#pragma unroll
            for (int c = 0; c < 2; c++)
                ap[c] = make_frag(*(const v8h*)&pr[c * 32 + kh * 16],
                                  *(const v8h*)&pr[c * 32 + kh * 16 + 8]);
        }
#pragma unroll
        for (int jd = 0; jd < 4; jd++) {
#pragma unroll
            for (int ch = 0; ch < 2; ch++) {
                const _Float16* vr = &Vl[cur][(jd * 16 + l16) * KSTR + ch * 32 + kh * 16];
                v16h bv = make_frag(*(const v8h*)&vr[0], *(const v8h*)&vr[8]);
                oacc[jd] = WMMA_F16(ap[ch], bv, oacc[jd]);
            }
        }
        __syncthreads();   // all waves done reading before buffer reuse
    }

    // normalize and write O in [B,T,C] f16 layout for the out-projection GEMM
    const int bb = bh >> 4, hh = bh & 15;
#pragma unroll
    for (int jd = 0; jd < 4; jd++) {
#pragma unroll
        for (int r = 0; r < 8; r++) {
            int m = kh * 8 + r;
            int token = (bb << 11) + qbase + m;
            float val = oacc[jd][r] / lrow[r];
            Ob[(size_t)token * DIMC + hh * HDIM + jd * 16 + l16] = (_Float16)val;
        }
    }
}

// ---------------------------------------------------------------------------
// Kernel 3: output projection  (out[n,j] = sum_c O[n,c]*W[j,c] + b[j], fp32)
// M=4096, N=1024, K=1024.  A (f16) double-buffered via async copies; B (f32)
// register-pipelined with f16 conversion.
// ---------------------------------------------------------------------------
__global__ __launch_bounds__(256) void out_proj_kernel(
    const _Float16* __restrict__ Ob, const float* __restrict__ W,
    const float* __restrict__ bias, float* __restrict__ out)
{
    __shared__ _Float16 As[2][128 * AS_STR] __attribute__((aligned(16)));
    __shared__ _Float16 Bs[64 * BS_STR] __attribute__((aligned(16)));

    const int tid  = threadIdx.x;
    const int lane = tid & 31, wave = tid >> 5;
    const int wm = wave >> 1, wn = wave & 1;
    const int l16 = lane & 15, kh = lane >> 4;
    const int Mbase = blockIdx.y * 128;
    const int Nbase = blockIdx.x * 64;

    float4 gB[2];
    auto issueA = [&](int kc, int buf) {   // 128x32 f16, 512 16B chunks, 2/thread
#pragma unroll
        for (int i = 0; i < 2; i++) {
            int f = tid + i * 256;
            int row = f >> 2, c8 = (f & 3) * 8;
            ASYNC_CP16(&Ob[(size_t)(Mbase + row) * DIMC + kc + c8],
                       &As[buf][row * AS_STR + c8]);
        }
    };
    auto loadB = [&](int kc) {
#pragma unroll
        for (int i = 0; i < 2; i++) {
            int f = tid + i * 256;
            int row = f >> 3, c4 = (f & 7) * 4;
            gB[i] = *(const float4*)&W[(size_t)(Nbase + row) * DIMC + kc + c4];
        }
    };
    auto storeB = [&]() {
#pragma unroll
        for (int i = 0; i < 2; i++) {
            int f = tid + i * 256;
            int row = f >> 3, c4 = (f & 7) * 4;
            *(v4h*)&Bs[row * BS_STR + c4] = cvt4(gB[i]);
        }
    };

    v8f acc[2][2] = {};

    issueA(0, 0);
    loadB(0);
    for (int it = 0; it < DIMC / 32; it++) {
        const int cur = it & 1;
        storeB();                                    // B tile -> LDS (f16)
        if (it + 1 < DIMC / 32) {
            issueA((it + 1) * 32, cur ^ 1);          // prefetch next A tile
            WAIT_ASYNC(2);                           // current A tile landed
        } else {
            WAIT_ASYNC(0);
        }
        __syncthreads();                             // tiles visible WG-wide
        if (it + 1 < DIMC / 32) loadB((it + 1) * 32);   // overlap with WMMAs

        v16h af[2], bf[2];
#pragma unroll
        for (int a = 0; a < 2; a++) {
            const _Float16* ar = &As[cur][(wm * 32 + a * 16 + l16) * AS_STR];
            af[a] = make_frag(*(const v8h*)&ar[kh * 16], *(const v8h*)&ar[kh * 16 + 8]);
        }
#pragma unroll
        for (int bi = 0; bi < 2; bi++) {
            const _Float16* br = &Bs[(wn * 32 + bi * 16 + l16) * BS_STR + kh * 16];
            bf[bi] = make_frag(*(const v8h*)&br[0], *(const v8h*)&br[8]);
        }
#pragma unroll
        for (int a = 0; a < 2; a++)
#pragma unroll
            for (int bi = 0; bi < 2; bi++)
                acc[a][bi] = WMMA_F16(af[a], bf[bi], acc[a][bi]);
        __syncthreads();                             // reads done before reuse
    }

#pragma unroll
    for (int bi = 0; bi < 2; bi++) {
        int j = Nbase + wn * 32 + bi * 16 + l16;
        float bj = bias[j];
#pragma unroll
        for (int a = 0; a < 2; a++) {
#pragma unroll
            for (int r = 0; r < 8; r++) {
                int gm = Mbase + wm * 32 + a * 16 + kh * 8 + r;
                out[(size_t)gm * DIMC + j] = acc[a][bi][r] + bj;
            }
        }
    }
}

// ---------------------------------------------------------------------------
extern "C" void kernel_launch(void* const* d_in, const int* in_sizes, int n_in,
                              void* d_out, int out_size, void* d_ws, size_t ws_size,
                              hipStream_t stream) {
    (void)in_sizes; (void)n_in; (void)out_size; (void)ws_size;
    const float* x     = (const float*)d_in[0];
    const float* w_in  = (const float*)d_in[1];
    const float* b_in  = (const float*)d_in[2];
    const float* w_out = (const float*)d_in[3];
    const float* b_out = (const float*)d_in[4];
    float* out = (float*)d_out;

    const size_t perbuf = (size_t)2 * NHEAD * SEQ * HDIM;  // 4M f16 = 8MB each
    _Float16* Qb = (_Float16*)d_ws;
    _Float16* Kb = Qb + perbuf;
    _Float16* Vb = Kb + perbuf;                            // stored as [B,H,D,T]
    _Float16* Ob = Vb + perbuf;                            // [4096,1024] f16

    qkv_proj_kernel<<<dim3(48, 32), 256, 0, stream>>>(x, w_in, b_in, Qb, Kb, Vb);
    flash_attn_kernel<<<dim3(16, 32), 256, 0, stream>>>(Qb, Kb, Vb, Ob);
    out_proj_kernel<<<dim3(16, 32), 256, 0, stream>>>(Ob, w_out, b_out, out);
}
